// CausalConvolution_72627896975499
// MI455X (gfx1250) — matrix operand order repaired and checked by reading, not compiled, not run on hardware
//
#include <hip/hip_runtime.h>
#include <hip/hip_bf16.h>

typedef __bf16 bf16;
typedef bf16  v16bf __attribute__((ext_vector_type(16)));
typedef float v8f   __attribute__((ext_vector_type(8)));

#define DIM_B    4
#define DIM_S    4096
#define DIM_H    2048
#define DIM_CIN  4096
#define DIM_COUT 4096
#define DIM_KW   4
#define DIM_G    8
#define DIM_M    (DIM_B * DIM_S)       // 16384
#define CPG      (DIM_CIN / DIM_G)     // 512 channels per group

// Use gfx1250 async DMA global->LDS (ASYNCcnt) instead of load+ds_store.
#define USE_ASYNC_LDS 1

union Frag16 { v16bf v; uint4 u[2]; };

// ---------------------------------------------------------------------------
// gfx1250 async global->LDS b128 (ISA 15.18.3 op 98, tracked by ASYNCcnt).
// LDS byte address = low 32 bits of the generic pointer (LDS aperture
// truncates to addr[31:0]).
// ---------------------------------------------------------------------------
__device__ __forceinline__ void async_load_b128(const bf16* g, bf16* l) {
  unsigned lo = (unsigned)(unsigned long long)(uintptr_t)l;
  asm volatile("global_load_async_to_lds_b128 %0, %1, off"
               :: "v"(lo), "v"((unsigned long long)(uintptr_t)g)
               : "memory");
}
__device__ __forceinline__ void wait_async0() {
  asm volatile("s_wait_asynccnt 0x0" ::: "memory");
}
__device__ __forceinline__ void wait_async6() {
  asm volatile("s_wait_asynccnt 0x6" ::: "memory");
}

// ---------------------------------------------------------------------------
// fp32 -> bf16 conversion, 8 elements / thread, 16B stores
// ---------------------------------------------------------------------------
__global__ __launch_bounds__(256) void cvt_f32_to_bf16_kernel(
    const float* __restrict__ in, bf16* __restrict__ out, long n) {
  long i = ((long)blockIdx.x * blockDim.x + threadIdx.x) * 8;
  if (i >= n) return;
  float4 f0 = *reinterpret_cast<const float4*>(in + i);
  float4 f1 = *reinterpret_cast<const float4*>(in + i + 4);
  union { bf16 h[8]; uint4 u; } p;
  p.h[0] = (bf16)f0.x; p.h[1] = (bf16)f0.y; p.h[2] = (bf16)f0.z; p.h[3] = (bf16)f0.w;
  p.h[4] = (bf16)f1.x; p.h[5] = (bf16)f1.y; p.h[6] = (bf16)f1.z; p.h[7] = (bf16)f1.w;
  *reinterpret_cast<uint4*>(out + i) = p.u;
}

// ---------------------------------------------------------------------------
// conv_w [COUT, CPG, 4] fp32  ->  4 tap matrices Wk[tap][COUT][CPG] bf16
// ---------------------------------------------------------------------------
__global__ __launch_bounds__(256) void cvt_convw_kernel(
    const float* __restrict__ cw, bf16* __restrict__ Wk) {
  long idx = (long)blockIdx.x * blockDim.x + threadIdx.x;   // co*CPG + ci
  if (idx >= (long)DIM_COUT * CPG) return;
  float4 f = *reinterpret_cast<const float4*>(cw + idx * 4);
  const long tapStride = (long)DIM_COUT * CPG;
  Wk[0 * tapStride + idx] = (bf16)f.x;
  Wk[1 * tapStride + idx] = (bf16)f.y;
  Wk[2 * tapStride + idx] = (bf16)f.z;
  Wk[3 * tapStride + idx] = (bf16)f.w;
}

// ---------------------------------------------------------------------------
// Tiled bf16 WMMA GEMM:  C[M,N] = A[M,K] * W[N,K]^T + bias  (optional SiLU)
// Block tile 128x256, BK=32, 8 waves (2x4), 64x64 per wave -> 16 WMMA accums.
// Double-buffered LDS, async global->LDS DMA (ASYNCcnt) one tile ahead.
// ---------------------------------------------------------------------------
template <bool SILU, bool OUT_BF16>
__global__ __launch_bounds__(256) void wmma_gemm_kernel(
    const bf16* __restrict__ A, const bf16* __restrict__ W,
    const float* __restrict__ bias, void* __restrict__ Cout,
    int M, int N, int K) {
  constexpr int BM = 128, BN = 256, BK = 32, LDT = BK + 8;  // +8 pad: 80B row
  __shared__ __align__(16) bf16 sA[2][BM * LDT];
  __shared__ __align__(16) bf16 sB[2][BN * LDT];

  const int tid = threadIdx.x, lane = tid & 31, wave = tid >> 5;
  const int ln = lane & 15, lh = lane >> 4;
  const int waveM = wave >> 2, waveN = wave & 3;
  const long mBlock = (long)blockIdx.y * BM;
  const long nBlock = (long)blockIdx.x * BN;
  const int aRow = tid >> 1, aOff = (tid & 1) * 16;

  const v8f vzero = {0.f, 0.f, 0.f, 0.f, 0.f, 0.f, 0.f, 0.f};
  v8f acc[4][4];
#pragma unroll
  for (int i = 0; i < 4; i++)
#pragma unroll
    for (int j = 0; j < 4; j++) acc[i][j] = vzero;

  const bf16* aPtr = A + (mBlock + aRow) * (long)K + aOff;
  const bf16* bPtr = W + (nBlock + tid) * (long)K;

  auto issueTile = [&](int k0, int buf) {
    bf16* dA = &sA[buf][aRow * LDT + aOff];
    bf16* dB = &sB[buf][tid * LDT];
#if USE_ASYNC_LDS
    async_load_b128(aPtr + k0,      dA);
    async_load_b128(aPtr + k0 + 8,  dA + 8);
    async_load_b128(bPtr + k0,      dB);
    async_load_b128(bPtr + k0 + 8,  dB + 8);
    async_load_b128(bPtr + k0 + 16, dB + 16);
    async_load_b128(bPtr + k0 + 24, dB + 24);
#else
    *reinterpret_cast<uint4*>(dA)      = *reinterpret_cast<const uint4*>(aPtr + k0);
    *reinterpret_cast<uint4*>(dA + 8)  = *reinterpret_cast<const uint4*>(aPtr + k0 + 8);
    *reinterpret_cast<uint4*>(dB)      = *reinterpret_cast<const uint4*>(bPtr + k0);
    *reinterpret_cast<uint4*>(dB + 8)  = *reinterpret_cast<const uint4*>(bPtr + k0 + 8);
    *reinterpret_cast<uint4*>(dB + 16) = *reinterpret_cast<const uint4*>(bPtr + k0 + 16);
    *reinterpret_cast<uint4*>(dB + 24) = *reinterpret_cast<const uint4*>(bPtr + k0 + 24);
#endif
  };

  const int steps = K / BK;
  issueTile(0, 0);
  for (int s = 0; s < steps; s++) {
    const int buf = s & 1;
    const bool hasNext = (s + 1) < steps;
    if (hasNext) issueTile((s + 1) * BK, buf ^ 1);   // prefetch next tile (in flight)
#if USE_ASYNC_LDS
    if (hasNext) wait_async6();   // oldest 6 (current buffer) complete
    else         wait_async0();
#endif
    __syncthreads();

    Frag16 bfrag[4];
#pragma unroll
    for (int ni = 0; ni < 4; ni++) {  // B frag: lane = N, 16 contiguous K at lh*16
      const bf16* p = &sB[buf][(waveN * 64 + ni * 16 + ln) * LDT + lh * 16];
      bfrag[ni].u[0] = *reinterpret_cast<const uint4*>(p);
      bfrag[ni].u[1] = *reinterpret_cast<const uint4*>(p + 8);
    }
#pragma unroll
    for (int mi = 0; mi < 4; mi++) {  // A frag: lane = M, 16 contiguous K at lh*16
      Frag16 af;
      const bf16* pa = &sA[buf][(waveM * 64 + mi * 16 + ln) * LDT];
      af.u[0] = *reinterpret_cast<const uint4*>(pa + lh * 16);
      af.u[1] = *reinterpret_cast<const uint4*>(pa + lh * 16 + 8);
#pragma unroll
      for (int ni = 0; ni < 4; ni++)
        acc[mi][ni] = __builtin_amdgcn_wmma_f32_16x16x32_bf16(
            false, af.v, false, bfrag[ni].v, (short)0, acc[mi][ni], false, false);
    }
    __syncthreads();   // everyone done reading buf before it is overwritten
  }

#pragma unroll
  for (int ni = 0; ni < 4; ni++) {
    const long n = nBlock + waveN * 64 + ni * 16 + ln;
    const float bn = bias[n];
#pragma unroll
    for (int mi = 0; mi < 4; mi++) {
      const long mBase = mBlock + waveM * 64 + mi * 16 + lh * 8;
#pragma unroll
      for (int r = 0; r < 8; r++) {
        float v = acc[mi][ni][r] + bn;
        if (SILU) v = v / (1.f + __expf(-v));
        if (OUT_BF16) ((bf16*)Cout)[(mBase + r) * (long)N + n] = (bf16)v;
        else          ((float*)Cout)[(mBase + r) * (long)N + n] = v;
      }
    }
  }
}

// ---------------------------------------------------------------------------
// Causal grouped conv as 4 shifted WMMA GEMMs + fused bias + SiLU.
// y[b,s,co] = silu( sum_{tap,ci} Wk[tap][co][ci] * h[b, s+tap-3, g(co)*CPG+ci] + b[co] )
// Same double-buffered async pipeline; 64 flattened (tap,k0) steps.
// ---------------------------------------------------------------------------
__global__ __launch_bounds__(256) void wmma_conv_kernel(
    const bf16* __restrict__ Hm,   // [DIM_M, DIM_CIN] bf16
    const bf16* __restrict__ Wk,   // [4][DIM_COUT][CPG] bf16
    const float* __restrict__ bias,
    bf16* __restrict__ Y) {        // [DIM_M, DIM_COUT] bf16
  constexpr int BM = 128, BN = 256, BK = 32, LDT = BK + 8;
  __shared__ __align__(16) bf16 sA[2][BM * LDT];
  __shared__ __align__(16) bf16 sB[2][BN * LDT];

  const int tid = threadIdx.x, lane = tid & 31, wave = tid >> 5;
  const int ln = lane & 15, lh = lane >> 4;
  const int waveM = wave >> 2, waveN = wave & 3;
  const long mBlock = (long)blockIdx.y * BM;
  const long nBlock = (long)blockIdx.x * BN;   // stays inside one group (512 % 256 == 0)
  const long gbase = (nBlock / CPG) * CPG;
  const int aRow = tid >> 1, aOff = (tid & 1) * 16;

  const long mRow = mBlock + aRow;
  const int batch = (int)(mRow / DIM_S);
  const int sl = (int)(mRow % DIM_S);

  const v8f vzero = {0.f, 0.f, 0.f, 0.f, 0.f, 0.f, 0.f, 0.f};
  v8f acc[4][4];
#pragma unroll
  for (int i = 0; i < 4; i++)
#pragma unroll
    for (int j = 0; j < 4; j++) acc[i][j] = vzero;

  const uint4 z4 = make_uint4(0u, 0u, 0u, 0u);
  const long tapStride = (long)DIM_COUT * CPG;

  // flattened step: tap = step>>4, k0 = (step&15)*BK   (CPG/BK == 16 steps/tap)
  auto issueTile = [&](int step, int buf) {
    const int tap = step >> 4;
    const int k0 = (step & 15) * BK;
    const int ss = sl + tap - (DIM_KW - 1);   // causal left shift
    const bool valid = (ss >= 0);
    const bf16* aP = Hm + ((long)batch * DIM_S + ss) * DIM_CIN + gbase + aOff + k0;
    const bf16* bP = Wk + tap * tapStride + (nBlock + tid) * (long)CPG + k0;
    bf16* dA = &sA[buf][aRow * LDT + aOff];
    bf16* dB = &sB[buf][tid * LDT];
#if USE_ASYNC_LDS
    if (valid) {                // exec-masked DMA; ASYNCcnt still +2 per wave
      async_load_b128(aP,     dA);
      async_load_b128(aP + 8, dA + 8);
    } else {                    // causal zero-pad rows (DScnt path)
      *reinterpret_cast<uint4*>(dA)     = z4;
      *reinterpret_cast<uint4*>(dA + 8) = z4;
    }
    async_load_b128(bP,      dB);
    async_load_b128(bP + 8,  dB + 8);
    async_load_b128(bP + 16, dB + 16);
    async_load_b128(bP + 24, dB + 24);
#else
    uint4 a0 = z4, a1 = z4;
    if (valid) {
      a0 = *reinterpret_cast<const uint4*>(aP);
      a1 = *reinterpret_cast<const uint4*>(aP + 8);
    }
    *reinterpret_cast<uint4*>(dA)      = a0;
    *reinterpret_cast<uint4*>(dA + 8)  = a1;
    *reinterpret_cast<uint4*>(dB)      = *reinterpret_cast<const uint4*>(bP);
    *reinterpret_cast<uint4*>(dB + 8)  = *reinterpret_cast<const uint4*>(bP + 8);
    *reinterpret_cast<uint4*>(dB + 16) = *reinterpret_cast<const uint4*>(bP + 16);
    *reinterpret_cast<uint4*>(dB + 24) = *reinterpret_cast<const uint4*>(bP + 24);
#endif
  };

  const int steps = DIM_KW * (CPG / BK);   // 64
  issueTile(0, 0);
  for (int s = 0; s < steps; s++) {
    const int buf = s & 1;
    const bool hasNext = (s + 1) < steps;
    if (hasNext) issueTile(s + 1, buf ^ 1);
#if USE_ASYNC_LDS
    if (hasNext) wait_async6();
    else         wait_async0();
#endif
    __syncthreads();

    Frag16 bfrag[4];
#pragma unroll
    for (int ni = 0; ni < 4; ni++) {
      const bf16* p = &sB[buf][(waveN * 64 + ni * 16 + ln) * LDT + lh * 16];
      bfrag[ni].u[0] = *reinterpret_cast<const uint4*>(p);
      bfrag[ni].u[1] = *reinterpret_cast<const uint4*>(p + 8);
    }
#pragma unroll
    for (int mi = 0; mi < 4; mi++) {
      Frag16 af;
      const bf16* pa = &sA[buf][(waveM * 64 + mi * 16 + ln) * LDT];
      af.u[0] = *reinterpret_cast<const uint4*>(pa + lh * 16);
      af.u[1] = *reinterpret_cast<const uint4*>(pa + lh * 16 + 8);
#pragma unroll
      for (int ni = 0; ni < 4; ni++)
        acc[mi][ni] = __builtin_amdgcn_wmma_f32_16x16x32_bf16(
            false, af.v, false, bfrag[ni].v, (short)0, acc[mi][ni], false, false);
    }
    __syncthreads();
  }

#pragma unroll
  for (int ni = 0; ni < 4; ni++) {
    const long n = nBlock + waveN * 64 + ni * 16 + ln;
    const float bn = bias[n];
#pragma unroll
    for (int mi = 0; mi < 4; mi++) {
      const long mBase = mBlock + waveM * 64 + mi * 16 + lh * 8;
#pragma unroll
      for (int r = 0; r < 8; r++) {
        float v = acc[mi][ni][r] + bn;
        v = v / (1.f + __expf(-v));  // SiLU
        Y[(mBase + r) * (long)DIM_COUT + n] = (bf16)v;
      }
    }
  }
}

// ---------------------------------------------------------------------------
// Host launcher
// ---------------------------------------------------------------------------
extern "C" void kernel_launch(void* const* d_in, const int* in_sizes, int n_in,
                              void* d_out, int out_size, void* d_ws, size_t ws_size,
                              hipStream_t stream) {
  const float* x      = (const float*)d_in[0];   // [B,S,H]
  const float* W_in   = (const float*)d_in[1];   // [CIN,H]
  const float* b_in   = (const float*)d_in[2];   // [CIN]
  const float* conv_w = (const float*)d_in[3];   // [COUT,CPG,4]
  const float* conv_b = (const float*)d_in[4];   // [COUT]
  const float* W_out  = (const float*)d_in[5];   // [H,COUT]
  const float* b_out  = (const float*)d_in[6];   // [H]
  float* out = (float*)d_out;                    // [B,S,H] fp32

  char* ws = (char*)d_ws;
  // workspace layout (bytes): needs ~368 MiB
  bf16* h_bf   = (bf16*)(ws);                       // 128 MiB  [M, CIN]
  bf16* y_bf   = (bf16*)(ws + (size_t)134217728);   // 128 MiB  [M, COUT]
  bf16* x_bf   = (bf16*)(ws + (size_t)268435456);   //  64 MiB  [M, H]
  bf16* win_b  = (bf16*)(ws + (size_t)335544320);   //  16 MiB  [CIN, H]
  bf16* wout_b = (bf16*)(ws + (size_t)352321536);   //  16 MiB  [H, COUT]
  bf16* wk_b   = (bf16*)(ws + (size_t)369098752);   //  16 MiB  [4][COUT][CPG]

  // 1) precision conversion (bf16 operands for WMMA)
  const long nx  = (long)DIM_M * DIM_H;             // 33,554,432
  const long nwi = (long)DIM_CIN * DIM_H;           //  8,388,608
  const long nwo = (long)DIM_H * DIM_COUT;          //  8,388,608
  cvt_f32_to_bf16_kernel<<<(int)(nx / 8 / 256), 256, 0, stream>>>(x, x_bf, nx);
  cvt_f32_to_bf16_kernel<<<(int)(nwi / 8 / 256), 256, 0, stream>>>(W_in, win_b, nwi);
  cvt_f32_to_bf16_kernel<<<(int)(nwo / 8 / 256), 256, 0, stream>>>(W_out, wout_b, nwo);
  cvt_convw_kernel<<<(int)((long)DIM_COUT * CPG / 256), 256, 0, stream>>>(conv_w, wk_b);

  // 2) h = x * W_in^T + b_in            (M=16384, N=4096, K=2048) -> bf16
  wmma_gemm_kernel<false, true><<<dim3(DIM_CIN / 256, DIM_M / 128), 256, 0, stream>>>(
      x_bf, win_b, b_in, (void*)h_bf, DIM_M, DIM_CIN, DIM_H);

  // 3) y = silu(grouped causal conv(h) + conv_b)   -> bf16
  wmma_conv_kernel<<<dim3(DIM_COUT / 256, DIM_M / 128), 256, 0, stream>>>(
      h_bf, wk_b, conv_b, y_bf);

  // 4) out = y * W_out^T + b_out        (M=16384, N=2048, K=4096) -> fp32
  wmma_gemm_kernel<false, false><<<dim3(DIM_H / 256, DIM_M / 128), 256, 0, stream>>>(
      y_bf, wout_b, b_out, (void*)out, DIM_M, DIM_H, DIM_COUT);
}
